// DeltaLM_7524782702900
// MI455X (gfx1250) — compile-verified, matrix operand order checked by reading, not run on hardware
//
#include <hip/hip_runtime.h>

typedef __attribute__((ext_vector_type(16))) __bf16 v16bf;
typedef __attribute__((ext_vector_type(8)))  float  v8f;
typedef __attribute__((ext_vector_type(8)))  unsigned short ushort8;
typedef int i128v __attribute__((vector_size(16)));   // b128 payload type for async builtins

#define DEV __device__ __forceinline__

static constexpr int Bdim = 4, Tdim = 2048, Ddim = 512, Ldim = 3, Vdim = 256;
static constexpr int NDdim = 4 * Ddim;            // 2048
static constexpr int Mdim  = Bdim * Tdim;         // 8192
static constexpr int CH = 128, NCH = Tdim / CH;   // chunked time scans

DEV unsigned short f2bf(float f) {
  unsigned int u = __float_as_uint(f);
  unsigned int r = (u + 0x7FFFu + ((u >> 16) & 1u)) >> 16;  // RNE
  return (unsigned short)r;
}
DEV float softplusf(float x) { return log1pf(expf(x)); }

// --- CDNA5 async global->LDS copy (ASYNCcnt path), builtin with asm fallback ---
DEV void async_cp_b128(const void* g, void* l) {
#if __has_builtin(__builtin_amdgcn_global_load_async_to_lds_b128)
  __builtin_amdgcn_global_load_async_to_lds_b128(
      (__attribute__((address_space(1))) i128v*)(g),
      (__attribute__((address_space(3))) i128v*)(l), 0, 0);
#else
  asm volatile("global_load_async_to_lds_b128 %0, %1, off"
               :: "v"((unsigned)(size_t)(l)), "v"(g) : "memory");
#endif
}
#if __has_builtin(__builtin_amdgcn_s_wait_asynccnt)
#define WAIT_ASYNC(n) __builtin_amdgcn_s_wait_asynccnt(n)
#else
#define WAIT_ASYNC(n) asm volatile("s_wait_asynccnt %0" :: "i"(n) : "memory")
#endif

// ---------------------------------------------------------------------------
// Stage 1: h = embed[x]; part = tanh(cumprod_d(h)) (wave32 multiplicative scan)
// ---------------------------------------------------------------------------
__global__ void embed_cumprod_kernel(const int* __restrict__ x,
                                     const float* __restrict__ embed,
                                     float* __restrict__ h,
                                     float* __restrict__ part) {
  int lane = threadIdx.x & 31, w = threadIdx.x >> 5;
  int tok = blockIdx.x * 8 + w;
  int row = x[tok];
  const float* e = embed + (size_t)row * Ddim;
  float carry = 1.f;
  for (int c = 0; c < Ddim / 32; ++c) {
    int d = c * 32 + lane;
    float v = e[d];
    h[(size_t)tok * Ddim + d] = v;
    float p = v;
#pragma unroll
    for (int off = 1; off < 32; off <<= 1) {
      float o = __shfl_up(p, off, 32);
      if (lane >= off) p *= o;
    }
    p *= carry;
    carry = __shfl(p, 31, 32);
    part[(size_t)tok * Ddim + d] = tanhf(p);
  }
}

// ---------------------------------------------------------------------------
// Chunked time scan, phase A: run = a*run + src (a=1 => cumsum). dst may alias src.
// ---------------------------------------------------------------------------
__global__ void scanA_kernel(const float* __restrict__ src, float* __restrict__ dst,
                             float* __restrict__ carry,
                             const float* __restrict__ kvec, int l, int decay) {
  int d = threadIdx.x;
  int c = blockIdx.x, b = blockIdx.y;
  float a = decay ? expf(-softplusf(kvec[l])) : 1.f;
  float run = 0.f;
  size_t base = ((size_t)b * Tdim + (size_t)c * CH) * Ddim + d;
  for (int t = 0; t < CH; ++t) {
    run = a * run + src[base + (size_t)t * Ddim];
    dst[base + (size_t)t * Ddim] = run;
  }
  carry[((size_t)b * NCH + c) * Ddim + d] = run;
}

// Phase B: exclusive scan of chunk carries with factor a^CH (in place).
__global__ void scanB_kernel(float* __restrict__ carry,
                             const float* __restrict__ kvec, int l, int decay) {
  int d = threadIdx.x, b = blockIdx.x;
  float aCH = decay ? expf(-softplusf(kvec[l]) * (float)CH) : 1.f;
  float e = 0.f;
  for (int c = 0; c < NCH; ++c) {
    size_t idx = ((size_t)b * NCH + c) * Ddim + d;
    float t = carry[idx];
    carry[idx] = e;
    e = t + aCH * e;
  }
}

// h = h * (1 + bd), bd = part + chunk carry (cumsum offsets)
__global__ void bd_apply_kernel(float* __restrict__ h, const float* __restrict__ part,
                                const float* __restrict__ carry) {
  size_t idx = (size_t)blockIdx.x * blockDim.x + threadIdx.x;
  if (idx >= (size_t)Mdim * Ddim) return;
  int d = (int)(idx % Ddim);
  size_t tok = idx / Ddim;
  int t = (int)(tok % Tdim), b = (int)(tok / Tdim), c = t / CH;
  float bd = part[idx] + carry[((size_t)b * NCH + c) * Ddim + d];
  h[idx] = h[idx] * (1.f + bd);
}

// mag[b,t] = || cumsum_t(h) - h[t-1] ||_2  (wave per token)
__global__ void mag_kernel(const float* __restrict__ h, const float* __restrict__ part,
                           const float* __restrict__ carry, float* __restrict__ mag) {
  int lane = threadIdx.x & 31, w = threadIdx.x >> 5;
  int tok = blockIdx.x * 8 + w;
  int t = tok % Tdim, b = tok / Tdim, c = t / CH;
  float ss = 0.f;
#pragma unroll
  for (int i = 0; i < Ddim / 32; ++i) {
    int d = i * 32 + lane;
    float hc = part[(size_t)tok * Ddim + d] + carry[((size_t)b * NCH + c) * Ddim + d];
    float hp = (t > 0) ? h[((size_t)tok - 1) * Ddim + d] : 0.f;
    float df = hc - hp;
    ss += df * df;
  }
#pragma unroll
  for (int off = 16; off; off >>= 1) ss += __shfl_xor(ss, off, 32);
  if (lane == 0) mag[tok] = sqrtf(ss);
}

// ctx = |mag * (part + a^(tl+1)*carry)|; layernorm; -> bf16 A-matrix
__global__ void ctx_ln_kernel(const float* __restrict__ part, const float* __restrict__ carry,
                              const float* __restrict__ mag,
                              const float* __restrict__ gamma, const float* __restrict__ beta,
                              const float* __restrict__ kvec, int l,
                              unsigned short* __restrict__ outbf) {
  int lane = threadIdx.x & 31, w = threadIdx.x >> 5;
  int tok = blockIdx.x * 8 + w;
  int t = tok % Tdim, b = tok / Tdim, c = t / CH, tl = t % CH;
  float sp = softplusf(kvec[l]);
  float scale = expf(-sp * (float)(tl + 1));
  float m = mag[tok];
  float r[Ddim / 32];
  float s = 0.f, s2 = 0.f;
#pragma unroll
  for (int i = 0; i < Ddim / 32; ++i) {
    int d = i * 32 + lane;
    float v = part[(size_t)tok * Ddim + d] + scale * carry[((size_t)b * NCH + c) * Ddim + d];
    v = fabsf(m * v);
    r[i] = v;
    s += v; s2 += v * v;
  }
#pragma unroll
  for (int off = 16; off; off >>= 1) { s += __shfl_xor(s, off, 32); s2 += __shfl_xor(s2, off, 32); }
  float mu = s / (float)Ddim;
  float var = s2 / (float)Ddim - mu * mu;
  float rs = rsqrtf(var + 1e-3f);
#pragma unroll
  for (int i = 0; i < Ddim / 32; ++i) {
    int d = i * 32 + lane;
    float y = (r[i] - mu) * rs * gamma[d] + beta[d];
    outbf[(size_t)tok * Ddim + d] = f2bf(y);
  }
}

// f32 (R x C) -> bf16 transposed (C x R), LDS-tiled
__global__ void transpose_bf16_kernel(const float* __restrict__ in,
                                      unsigned short* __restrict__ out, int R, int Cc) {
  __shared__ float tile[32][33];
  int c0 = blockIdx.x * 32, r0 = blockIdx.y * 32;
  int tx = threadIdx.x, ty = threadIdx.y;  // block (32,8)
#pragma unroll
  for (int j = 0; j < 32; j += 8)
    tile[ty + j][tx] = in[(size_t)(r0 + ty + j) * Cc + c0 + tx];
  __syncthreads();
#pragma unroll
  for (int j = 0; j < 32; j += 8)
    out[(size_t)(c0 + ty + j) * R + r0 + tx] = f2bf(tile[tx][ty + j]);
}

__global__ void cvt_bf16_kernel(const float* __restrict__ in,
                                unsigned short* __restrict__ out, size_t n) {
  size_t i = (size_t)blockIdx.x * blockDim.x + threadIdx.x;
  if (i < n) out[i] = f2bf(in[i]);
}

// ---------------------------------------------------------------------------
// WMMA bf16 GEMM, LDS double-buffered via async global->LDS copies.
// C = A(MxK) * Bt(NxK)^T, f32 accumulate. Block: 256 thr (8 waves), tile 128x128,
// wave tile 64x32, K-step 32. LDS row stride padded to 40 ushorts (80B): the 16
// fragment rows map to 16 distinct 4-bank groups -> conflict-free ds_load_b128.
// EPI 0: Cbf = bf16(gelu(C+bias)); EPI 1: Cf += C+bias; EPI 2: Cf = C.
// ---------------------------------------------------------------------------
static constexpr int LDSW = 40;  // padded row stride (ushorts)

DEV v16bf lds_frag(const unsigned short* p) {
  union { v16bf v; ushort8 h[2]; } u;
  u.h[0] = *(const ushort8*)(p);        // k0 .. k0+7
  u.h[1] = *(const ushort8*)(p + 16);   // k0+16 .. k0+23
  return u.v;
}

template <int EPI>
__global__ __launch_bounds__(256) void gemm_bf16_kernel(
    const unsigned short* __restrict__ A,   // M x K bf16
    const unsigned short* __restrict__ Bt,  // N x K bf16 (pre-transposed weight)
    const float* __restrict__ bias,         // N (EPI 0/1)
    float* __restrict__ Cf,                 // EPI 1 (residual +=) / EPI 2 (store)
    unsigned short* __restrict__ Cbf,       // EPI 0 (bf16 store)
    int Nn, int Kn) {
  __shared__ __align__(16) unsigned short sA[2][128 * LDSW];
  __shared__ __align__(16) unsigned short sB[2][128 * LDSW];

  int tid = threadIdx.x;
  int lane = tid & 31, w = tid >> 5;
  int wm = w >> 2, wn = w & 3;            // 2x4 wave grid
  int rA = lane & 15;
  int koff = (lane & 16) ? 8 : 0;         // ISA 16-bit fragment K split

  // async copy mapping: thread covers 8-element chunks tid and tid+256 of each tile
  int r0 = tid >> 2, c0 = (tid & 3) * 8;  // rows 0..63 / cols {0,8,16,24}
  int r1 = r0 + 64;                       // rows 64..127
  const unsigned short* gA0 = A  + (size_t)(blockIdx.y * 128 + r0) * Kn + c0;
  const unsigned short* gA1 = A  + (size_t)(blockIdx.y * 128 + r1) * Kn + c0;
  const unsigned short* gB0 = Bt + (size_t)(blockIdx.x * 128 + r0) * Kn + c0;
  const unsigned short* gB1 = Bt + (size_t)(blockIdx.x * 128 + r1) * Kn + c0;
  unsigned lA0 = r0 * LDSW + c0, lA1 = r1 * LDSW + c0;  // ushort offsets (16B aligned)

  auto issue = [&](int st, int kk) {
    async_cp_b128(gA0 + kk, &sA[st][lA0]);
    async_cp_b128(gA1 + kk, &sA[st][lA1]);
    async_cp_b128(gB0 + kk, &sB[st][lA0]);
    async_cp_b128(gB1 + kk, &sB[st][lA1]);
  };

  v8f acc[4][2] = {};
  issue(0, 0);
  for (int kk = 0; kk < Kn; kk += 32) {
    int cur = (kk >> 5) & 1;
    if (kk + 32 < Kn) {
      issue(cur ^ 1, kk + 32);  // prefetch next stage
      WAIT_ASYNC(4);            // previous 4 (current stage) complete, 4 in flight
    } else {
      WAIT_ASYNC(0);
    }
    __syncthreads();            // all waves' stage data visible

    v16bf af[4], bfr[2];
#pragma unroll
    for (int i = 0; i < 4; ++i)
      af[i] = lds_frag(&sA[cur][(wm * 64 + i * 16 + rA) * LDSW + koff]);
#pragma unroll
    for (int j = 0; j < 2; ++j)
      bfr[j] = lds_frag(&sB[cur][(wn * 32 + j * 16 + rA) * LDSW + koff]);
#pragma unroll
    for (int i = 0; i < 4; ++i)
#pragma unroll
      for (int j = 0; j < 2; ++j)
        acc[i][j] = __builtin_amdgcn_wmma_f32_16x16x32_bf16(
            false, af[i], false, bfr[j], (short)0, acc[i][j], false, false);
    __syncthreads();            // reads done before this buffer is re-filled
  }

  int rowoff = (lane & 16) ? 8 : 0;
  int col = lane & 15;
  int m0 = blockIdx.y * 128 + wm * 64;
  int n0 = blockIdx.x * 128 + wn * 32;
#pragma unroll
  for (int i = 0; i < 4; ++i)
#pragma unroll
    for (int j = 0; j < 2; ++j)
#pragma unroll
      for (int e = 0; e < 8; ++e) {
        int mrow = m0 + i * 16 + e + rowoff;
        int ncol = n0 + j * 16 + col;
        float v = acc[i][j][e];
        if (EPI == 0) {
          v += bias[ncol];
          v = 0.5f * v * (1.f + erff(v * 0.70710678118f));  // exact gelu
          Cbf[(size_t)mrow * Nn + ncol] = f2bf(v);
        } else if (EPI == 1) {
          v += bias[ncol];
          Cf[(size_t)mrow * Nn + ncol] += v;                // residual add
        } else {
          Cf[(size_t)mrow * Nn + ncol] = v;                 // logits
        }
      }
}

// ---------------------------------------------------------------------------
extern "C" void kernel_launch(void* const* d_in, const int* in_sizes, int n_in,
                              void* d_out, int out_size, void* d_ws, size_t ws_size,
                              hipStream_t stream) {
  const int*   x     = (const int*)d_in[0];
  const float* embed = (const float*)d_in[1];
  const float* k     = (const float*)d_in[2];
  const float* ln_g  = (const float*)d_in[3];
  const float* ln_b  = (const float*)d_in[4];
  const float* W1    = (const float*)d_in[5];
  const float* b1    = (const float*)d_in[6];
  const float* W2    = (const float*)d_in[7];
  const float* b2    = (const float*)d_in[8];
  float* out = (float*)d_out;

  char* ws = (char*)d_ws;
  size_t off = 0;
  auto alloc = [&](size_t bytes) {
    size_t o = (off + 255) & ~(size_t)255;
    off = o + bytes;
    return (void*)(ws + o);
  };
  float* h     = (float*)alloc((size_t)Mdim * Ddim * 4);
  float* part  = (float*)alloc((size_t)Mdim * Ddim * 4);
  float* carry = (float*)alloc((size_t)Bdim * NCH * Ddim * 4);
  float* mag   = (float*)alloc((size_t)Mdim * 4);
  unsigned short* Abf = (unsigned short*)alloc((size_t)Mdim * Ddim * 2);
  unsigned short* Gbf = (unsigned short*)alloc((size_t)Mdim * NDdim * 2);
  unsigned short* W1b = (unsigned short*)alloc((size_t)Ldim * NDdim * Ddim * 2);
  unsigned short* W2b = (unsigned short*)alloc((size_t)Ldim * Ddim * NDdim * 2);
  unsigned short* Eb  = (unsigned short*)alloc((size_t)Vdim * Ddim * 2);

  // weight prep: bf16 + transpose to (N x K)
  dim3 tb(32, 8);
  for (int l = 0; l < Ldim; ++l) {
    transpose_bf16_kernel<<<dim3(NDdim / 32, Ddim / 32), tb, 0, stream>>>(
        W1 + (size_t)l * Ddim * NDdim, W1b + (size_t)l * NDdim * Ddim, Ddim, NDdim);
    transpose_bf16_kernel<<<dim3(Ddim / 32, NDdim / 32), tb, 0, stream>>>(
        W2 + (size_t)l * NDdim * Ddim, W2b + (size_t)l * Ddim * NDdim, NDdim, Ddim);
  }
  cvt_bf16_kernel<<<(Vdim * Ddim + 255) / 256, 256, 0, stream>>>(embed, Eb,
                                                                 (size_t)Vdim * Ddim);

  // stage 1: embedding + boundary feature
  embed_cumprod_kernel<<<Mdim / 8, 256, 0, stream>>>(x, embed, h, part);
  scanA_kernel<<<dim3(NCH, Bdim), Ddim, 0, stream>>>(part, part, carry, k, 0, 0);
  scanB_kernel<<<Bdim, Ddim, 0, stream>>>(carry, k, 0, 0);
  bd_apply_kernel<<<(Mdim * Ddim + 255) / 256, 256, 0, stream>>>(h, part, carry);

  for (int l = 0; l < Ldim; ++l) {
    // DeltaImportance magnitude (time cumsum of h)
    scanA_kernel<<<dim3(NCH, Bdim), Ddim, 0, stream>>>(h, part, carry, k, l, 0);
    scanB_kernel<<<Bdim, Ddim, 0, stream>>>(carry, k, l, 0);
    mag_kernel<<<Mdim / 8, 256, 0, stream>>>(h, part, carry, mag);
    // decay context as first-order IIR scan (replaces TxT matmul)
    scanA_kernel<<<dim3(NCH, Bdim), Ddim, 0, stream>>>(h, part, carry, k, l, 1);
    scanB_kernel<<<Bdim, Ddim, 0, stream>>>(carry, k, l, 1);
    ctx_ln_kernel<<<Mdim / 8, 256, 0, stream>>>(part, carry, mag,
                                                ln_g + (size_t)l * Ddim,
                                                ln_b + (size_t)l * Ddim, k, l, Abf);
    // MLP: WMMA bf16 GEMMs (LDS double-buffered, async copies)
    gemm_bf16_kernel<0><<<dim3(NDdim / 128, Mdim / 128), 256, 0, stream>>>(
        Abf, W1b + (size_t)l * NDdim * Ddim, b1 + (size_t)l * NDdim,
        nullptr, Gbf, NDdim, Ddim);
    gemm_bf16_kernel<1><<<dim3(Ddim / 128, Mdim / 128), 256, 0, stream>>>(
        Gbf, W2b + (size_t)l * Ddim * NDdim, b2 + (size_t)l * Ddim,
        h, nullptr, Ddim, NDdim);
  }

  // tied output projection: logits = h @ embed^T (embed is already N x K bf16)
  cvt_bf16_kernel<<<(int)(((size_t)Mdim * Ddim + 255) / 256), 256, 0, stream>>>(
      h, Abf, (size_t)Mdim * Ddim);
  gemm_bf16_kernel<2><<<dim3(Vdim / 128, Mdim / 128), 256, 0, stream>>>(
      Abf, Eb, nullptr, out, nullptr, Vdim, Ddim);
}
